// AttnDecoderRNN_71442486001717
// MI455X (gfx1250) — matrix operand order repaired and checked by reading, not compiled, not run on hardware
//
#include <hip/hip_runtime.h>
#include <math.h>

typedef __attribute__((ext_vector_type(16))) __bf16 v16bf;
typedef __attribute__((ext_vector_type(8)))  __bf16 v8bf;
typedef __attribute__((ext_vector_type(8)))  float  v8f;

#define B_  64
#define P_  196
#define E_  2048
#define D_  512
#define A_  512
#define V_  10000
#define L_  20

// ---------------- bf16 fragment loaders (ISA 7.12.2 layouts) ----------------
// A 16x32 bf16: lane L -> row M=L%16; elems 0..7 = K kb+8*(L/16)+0..7,
// elems 8..15 = +16.  Two aligned b128 loads.
__device__ __forceinline__ v16bf load_a_bf(const __bf16* __restrict__ A, int lda,
                                           int mRow0, int kb, int lane) {
    const __bf16* ap = A + (size_t)(mRow0 + (lane & 15)) * lda + kb + ((lane >> 4) << 3);
    v8bf lo = *(const v8bf*)ap;
    v8bf hi = *(const v8bf*)(ap + 16);
    return __builtin_shufflevector(lo, hi, 0,1,2,3,4,5,6,7,8,9,10,11,12,13,14,15);
}
// B 32x16 with B[k][n] = W[n][k]: lane L -> col N=L%16;
// elems 0..7 = K kb+8*(L/16)+0..7, elems 8..15 = +16 (same as A). Two b128 loads.
__device__ __forceinline__ v16bf load_b_bf(const __bf16* __restrict__ W, int ldw,
                                           int nCol0, int kb, int lane) {
    const __bf16* wp = W + (size_t)(nCol0 + (lane & 15)) * ldw + kb + ((lane >> 4) << 3);
    v8bf lo = *(const v8bf*)wp;
    v8bf hi = *(const v8bf*)(wp + 16);
    __builtin_prefetch((const void*)(wp + 32), 0, 1);   // next K-tile -> global_prefetch_b8
    return __builtin_shufflevector(lo, hi, 0,1,2,3,4,5,6,7,8,9,10,11,12,13,14,15);
}

// ---- One K-stream of the GEMM, 2x-unrolled ping-pong double buffer.
// K must be a multiple of 64 (here: 512 or 2048). Buffers are reloaded in
// place (no register rotation copies) and the last pair is peeled so the
// loop body has unconditional loads -> branch-free, mov-free inner loop. ----
template<bool TAIL16>
__device__ __forceinline__ void gemm_k_stream(const __bf16* __restrict__ A,
                                              const __bf16* __restrict__ W, int K,
                                              int mRow0, int nCol0, int lane,
                                              v8f& acc0, v8f& acc1)
{
    v16bf a0  = load_a_bf(A, K, mRow0, 0, lane);
    v16bf b00 = load_b_bf(W, K, nCol0, 0, lane);
    v16bf b10 = b00;
    if (!TAIL16) b10 = load_b_bf(W, K, nCol0 + 16, 0, lane);

    v16bf a1, b01, b11;
    for (int kb = 0; kb + 64 < K; kb += 64) {
        a1  = load_a_bf(A, K, mRow0, kb + 32, lane);
        b01 = load_b_bf(W, K, nCol0, kb + 32, lane);
        if (!TAIL16) b11 = load_b_bf(W, K, nCol0 + 16, kb + 32, lane);
        acc0 = __builtin_amdgcn_wmma_f32_16x16x32_bf16(false, a0, false, b00,
                                                       (short)0, acc0, false, false);
        if (!TAIL16)
            acc1 = __builtin_amdgcn_wmma_f32_16x16x32_bf16(false, a0, false, b10,
                                                           (short)0, acc1, false, false);
        a0  = load_a_bf(A, K, mRow0, kb + 64, lane);
        b00 = load_b_bf(W, K, nCol0, kb + 64, lane);
        if (!TAIL16) b10 = load_b_bf(W, K, nCol0 + 16, kb + 64, lane);
        acc0 = __builtin_amdgcn_wmma_f32_16x16x32_bf16(false, a1, false, b01,
                                                       (short)0, acc0, false, false);
        if (!TAIL16)
            acc1 = __builtin_amdgcn_wmma_f32_16x16x32_bf16(false, a1, false, b11,
                                                           (short)0, acc1, false, false);
    }
    // peeled last pair (kb = K-64, K-32)
    a1  = load_a_bf(A, K, mRow0, K - 32, lane);
    b01 = load_b_bf(W, K, nCol0, K - 32, lane);
    if (!TAIL16) b11 = load_b_bf(W, K, nCol0 + 16, K - 32, lane);
    acc0 = __builtin_amdgcn_wmma_f32_16x16x32_bf16(false, a0, false, b00,
                                                   (short)0, acc0, false, false);
    if (!TAIL16)
        acc1 = __builtin_amdgcn_wmma_f32_16x16x32_bf16(false, a0, false, b10,
                                                       (short)0, acc1, false, false);
    acc0 = __builtin_amdgcn_wmma_f32_16x16x32_bf16(false, a1, false, b01,
                                                   (short)0, acc0, false, false);
    if (!TAIL16)
        acc1 = __builtin_amdgcn_wmma_f32_16x16x32_bf16(false, a1, false, b11,
                                                       (short)0, acc1, false, false);
}

// ---- Generic GEMM: C = act(A1@W1^T + b1 [+ A2@W2^T + b2]).
// 4 waves/block, each wave: 16(M)x32(N) tile (two accumulators sharing the
// A fragments). TAIL16 instance does a single 16-wide tile (N=10000 edge). ----
template<bool HASA2, bool TAIL16, int ACT, bool MASK>
__global__ __launch_bounds__(128)
void gemm_wmma(const __bf16* __restrict__ A1, const __bf16* __restrict__ W1,
               const float* __restrict__ b1, int K1,
               const __bf16* __restrict__ A2, const __bf16* __restrict__ W2,
               const float* __restrict__ b2, int K2,
               float* __restrict__ C, __bf16* __restrict__ Cbf, int ldc,
               int nOffset, const int* __restrict__ caplen, int tstep)
{
    const int lane  = threadIdx.x & 31;
    const int wave  = threadIdx.x >> 5;
    const int mRow0 = blockIdx.x * 64 + wave * 16;
    const int nCol0 = nOffset + blockIdx.y * 32;

    v8f acc0 = {}, acc1 = {};
    gemm_k_stream<TAIL16>(A1, W1, K1, mRow0, nCol0, lane, acc0, acc1);
    if (HASA2)
        gemm_k_stream<TAIL16>(A2, W2, K2, mRow0, nCol0, lane, acc0, acc1);

#pragma unroll
    for (int j = 0; j < (TAIL16 ? 1 : 2); ++j) {
        const int n = nCol0 + 16 * j + (lane & 15);
        const float bias = b1[n] + (HASA2 ? b2[n] : 0.f);
        const v8f acc = j ? acc1 : acc0;
#pragma unroll
        for (int r = 0; r < 8; ++r) {
            // C/D layout: lanes 0-15: M=r, N=lane; lanes 16-31: M=r+8, N=lane-16
            const int m = mRow0 + r + ((lane >> 4) << 3);
            float v = acc[r] + bias;
            if (ACT == 1) v = 1.f / (1.f + __expf(-v));                 // sigmoid
            if (MASK)     v *= (caplen[m] - 1 > tstep) ? 1.f : 0.f;     // row mask
            if (C)   C[(size_t)m * ldc + n]   = v;
            if (Cbf) Cbf[(size_t)m * ldc + n] = (__bf16)v;
        }
    }
}

// ---- f32 -> bf16 bulk conversion ----
__global__ __launch_bounds__(256)
void cvt_bf16(const float* __restrict__ src, __bf16* __restrict__ dst, int n)
{
    const int i = blockIdx.x * 256 + threadIdx.x;
    if (i < n) dst[i] = (__bf16)src[i];
}

// ---- e = relu(attn1 + attn2)·wl + bl ; softmax over P ; alpha (+masked out) ----
__global__ __launch_bounds__(256)
void attn_softmax(const float* __restrict__ attn1, const float* __restrict__ attn2,
                  const float* __restrict__ wl, const float* __restrict__ bl,
                  float* __restrict__ alpha, float* __restrict__ alphas_out,
                  const int* __restrict__ caplen, int tstep, int T)
{
    __shared__ float s_a2[A_];
    __shared__ float s_r[256];
    const int b = blockIdx.x;
    const int tid = threadIdx.x;

    for (int a = tid; a < A_; a += 256) s_a2[a] = attn2[b * A_ + a];
    __syncthreads();

    float e = -INFINITY;
    if (tid < P_) {
        const float* a1 = attn1 + ((size_t)b * P_ + tid) * A_;
        float s = 0.f;
        for (int a = 0; a < A_; ++a) {
            float v = a1[a] + s_a2[a];
            v = v > 0.f ? v : 0.f;
            s = fmaf(v, wl[a], s);
        }
        e = s + bl[0];
    }
    s_r[tid] = e;
    __syncthreads();
    for (int off = 128; off > 0; off >>= 1) {
        if (tid < off) s_r[tid] = fmaxf(s_r[tid], s_r[tid + off]);
        __syncthreads();
    }
    const float mx = s_r[0];
    __syncthreads();
    const float ex = (tid < P_) ? __expf(e - mx) : 0.f;
    s_r[tid] = ex;
    __syncthreads();
    for (int off = 128; off > 0; off >>= 1) {
        if (tid < off) s_r[tid] += s_r[tid + off];
        __syncthreads();
    }
    const float inv = 1.f / s_r[0];
    if (tid < P_) {
        const float al = ex * inv;
        alpha[b * P_ + tid] = al;
        const float m = (caplen[b] - 1 > tstep) ? 1.f : 0.f;
        alphas_out[((size_t)b * T + tstep) * P_ + tid] = al * m;
    }
}

// ---- ctx[b,e] = sum_p alpha[b,p] * enc[b,p,e] ----
__global__ __launch_bounds__(256)
void ctx_kernel(const float* __restrict__ alpha, const float* __restrict__ enc,
                float* __restrict__ ctx)
{
    const int b = blockIdx.y;
    const int e = blockIdx.x * 256 + threadIdx.x;
    const float* ep = enc + (size_t)b * P_ * E_ + e;
    const float* al = alpha + b * P_;
    float s = 0.f;
    for (int p = 0; p < P_; ++p) s = fmaf(al[p], ep[(size_t)p * E_], s);
    ctx[(size_t)b * E_ + e] = s;
}

// ---- x_t = emb[captions[b,t]] + gate * ctx  (bf16 out: only a GEMM reads it) ----
__global__ __launch_bounds__(256)
void xt_kernel(const int* __restrict__ captions, int tstep,
               const float* __restrict__ emb, const float* __restrict__ gate,
               const float* __restrict__ ctx, __bf16* __restrict__ xt_bf)
{
    const int b = blockIdx.y;
    const int e = blockIdx.x * 256 + threadIdx.x;
    const int tok = captions[b * L_ + tstep];
    const size_t i = (size_t)b * E_ + e;
    xt_bf[i] = (__bf16)(emb[(size_t)tok * E_ + e] + gate[i] * ctx[i]);
}

// ---- LSTM cell: c stays f32, h written as bf16 (only GEMMs consume h) ----
__global__ __launch_bounds__(256)
void lstm_kernel(const float* __restrict__ gates, __bf16* __restrict__ h_bf,
                 float* __restrict__ c)
{
    const int idx = blockIdx.x * 256 + threadIdx.x;  // B*D
    const int b = idx / D_, d = idx % D_;
    const float* g = gates + (size_t)b * (4 * D_);
    const float gi = g[d], gf = g[D_ + d], gg = g[2 * D_ + d], go = g[3 * D_ + d];
    const float si = 1.f / (1.f + __expf(-gi));
    const float sf = 1.f / (1.f + __expf(-gf));
    const float so = 1.f / (1.f + __expf(-go));
    const float cn = sf * c[idx] + si * tanhf(gg);
    const float hn = so * tanhf(cn);
    c[idx] = cn;
    h_bf[idx] = (__bf16)hn;
}

// ---- mean over P (bf16 out: only h0/c0 GEMMs consume it) ----
__global__ __launch_bounds__(256)
void mean_kernel(const float* __restrict__ enc, __bf16* __restrict__ meanE_bf)
{
    const int b = blockIdx.y;
    const int e = blockIdx.x * 256 + threadIdx.x;
    const float* ep = enc + (size_t)b * P_ * E_ + e;
    float s = 0.f;
    for (int p = 0; p < P_; ++p) s += ep[(size_t)p * E_];
    meanE_bf[(size_t)b * E_ + e] = (__bf16)(s * (1.f / (float)P_));
}

// ---- pass-through integer outputs (as f32) ----
__global__ __launch_bounds__(256)
void intout_kernel(const int* __restrict__ captions, const int* __restrict__ caplen,
                   float* __restrict__ out_cap, float* __restrict__ out_len)
{
    const int i = blockIdx.x * 256 + threadIdx.x;
    if (i < B_ * L_) out_cap[i] = (float)captions[i];
    if (i < B_)      out_len[i] = (float)(caplen[i] - 1);
}

extern "C" void kernel_launch(void* const* d_in, const int* in_sizes, int n_in,
                              void* d_out, int out_size, void* d_ws, size_t ws_size,
                              hipStream_t stream)
{
    (void)in_sizes; (void)n_in; (void)ws_size;
    const float* enc      = (const float*)d_in[0];
    const int*   captions = (const int*)d_in[1];
    const int*   caplen   = (const int*)d_in[2];
    const float* We  = (const float*)d_in[3];
    const float* be  = (const float*)d_in[4];
    const float* Wd  = (const float*)d_in[5];
    const float* bd  = (const float*)d_in[6];
    const float* wl  = (const float*)d_in[7];
    const float* bl  = (const float*)d_in[8];
    const float* emb = (const float*)d_in[9];
    const float* Wih = (const float*)d_in[10];
    const float* bih = (const float*)d_in[11];
    const float* Whh = (const float*)d_in[12];
    const float* bhh = (const float*)d_in[13];
    const float* Wh0 = (const float*)d_in[14];
    const float* bh0 = (const float*)d_in[15];
    const float* Wc0 = (const float*)d_in[16];
    const float* bc0 = (const float*)d_in[17];
    const float* Wb  = (const float*)d_in[18];
    const float* bb  = (const float*)d_in[19];
    const float* Wfc = (const float*)d_in[20];
    const float* bfc = (const float*)d_in[21];

    float* out = (float*)d_out;
    const int T = (out_size - B_ * L_ - B_) / (B_ * (V_ + P_));   // == 19
    float* out_pred   = out;
    float* out_cap    = out + (size_t)B_ * T * V_;
    float* out_len    = out_cap + (size_t)B_ * L_;
    float* out_alphas = out_len + B_;

    // -------- carve workspace (256B aligned chunks) --------
    char* wsb = (char*)d_ws;
    auto carve = [&](size_t bytes) -> void* {
        void* p = (void*)wsb;
        wsb += (bytes + 255) & ~(size_t)255;
        return p;
    };
    __bf16* enc_bf  = (__bf16*)carve((size_t)B_ * P_ * E_ * 2);
    __bf16* We_bf   = (__bf16*)carve((size_t)A_ * E_ * 2);
    __bf16* Wd_bf   = (__bf16*)carve((size_t)A_ * D_ * 2);
    __bf16* Wih_bf  = (__bf16*)carve((size_t)4 * D_ * E_ * 2);
    __bf16* Whh_bf  = (__bf16*)carve((size_t)4 * D_ * D_ * 2);
    __bf16* Wh0_bf  = (__bf16*)carve((size_t)D_ * E_ * 2);
    __bf16* Wc0_bf  = (__bf16*)carve((size_t)D_ * E_ * 2);
    __bf16* Wb_bf   = (__bf16*)carve((size_t)E_ * D_ * 2);
    __bf16* Wfc_bf  = (__bf16*)carve((size_t)V_ * D_ * 2);
    __bf16* meanE_bf= (__bf16*)carve((size_t)B_ * E_ * 2);
    __bf16* h_bf    = (__bf16*)carve((size_t)B_ * D_ * 2);
    __bf16* xt_bf   = (__bf16*)carve((size_t)B_ * E_ * 2);
    float*  attn1   = (float*)carve((size_t)B_ * P_ * A_ * 4);
    float*  c       = (float*)carve((size_t)B_ * D_ * 4);
    float*  attn2   = (float*)carve((size_t)B_ * D_ * 4);
    float*  alpha   = (float*)carve((size_t)B_ * P_ * 4);
    float*  ctx     = (float*)carve((size_t)B_ * E_ * 4);
    float*  gate    = (float*)carve((size_t)B_ * E_ * 4);
    float*  gates   = (float*)carve((size_t)B_ * 4 * D_ * 4);

    // -------- one-time f32 -> bf16 staging (L2-resident afterwards) --------
    auto cvt = [&](const float* s, __bf16* d, int n) {
        cvt_bf16<<<(n + 255) / 256, 256, 0, stream>>>(s, d, n);
    };
    cvt(enc, enc_bf, B_ * P_ * E_);
    cvt(We,  We_bf,  A_ * E_);
    cvt(Wd,  Wd_bf,  A_ * D_);
    cvt(Wih, Wih_bf, 4 * D_ * E_);
    cvt(Whh, Whh_bf, 4 * D_ * D_);
    cvt(Wh0, Wh0_bf, D_ * E_);
    cvt(Wc0, Wc0_bf, D_ * E_);
    cvt(Wb,  Wb_bf,  E_ * D_);
    cvt(Wfc, Wfc_bf, V_ * D_);

    // pass-through outputs
    intout_kernel<<<(B_ * L_ + 255) / 256, 256, 0, stream>>>(captions, caplen,
                                                             out_cap, out_len);
    // mean_enc, h0 (bf16 only), c0 (f32)
    mean_kernel<<<dim3(E_ / 256, B_), 256, 0, stream>>>(enc, meanE_bf);
    gemm_wmma<false,false,0,false><<<dim3(B_ / 64, D_ / 32), 128, 0, stream>>>(
        meanE_bf, Wh0_bf, bh0, E_, nullptr, nullptr, nullptr, 0,
        nullptr, h_bf, D_, 0, nullptr, 0);
    gemm_wmma<false,false,0,false><<<dim3(B_ / 64, D_ / 32), 128, 0, stream>>>(
        meanE_bf, Wc0_bf, bc0, E_, nullptr, nullptr, nullptr, 0,
        c, nullptr, D_, 0, nullptr, 0);
    // attn1 = enc_out(12544x2048) @ We^T + be
    gemm_wmma<false,false,0,false><<<dim3((B_ * P_) / 64, A_ / 32), 128, 0, stream>>>(
        enc_bf, We_bf, be, E_, nullptr, nullptr, nullptr, 0,
        attn1, nullptr, A_, 0, nullptr, 0);

    for (int t = 0; t < T; ++t) {
        // attn2 = h @ Wd^T + bd
        gemm_wmma<false,false,0,false><<<dim3(1, A_ / 32), 128, 0, stream>>>(
            h_bf, Wd_bf, bd, D_, nullptr, nullptr, nullptr, 0,
            attn2, nullptr, A_, 0, nullptr, 0);
        // e, softmax, alpha (+masked alphas out)
        attn_softmax<<<B_, 256, 0, stream>>>(attn1, attn2, wl, bl, alpha,
                                             out_alphas, caplen, t, T);
        // ctx = alpha @ enc_out   (f32 reduction over P)
        ctx_kernel<<<dim3(E_ / 256, B_), 256, 0, stream>>>(alpha, enc, ctx);
        // gate = sigmoid(h @ Wb^T + bb)
        gemm_wmma<false,false,1,false><<<dim3(1, E_ / 32), 128, 0, stream>>>(
            h_bf, Wb_bf, bb, D_, nullptr, nullptr, nullptr, 0,
            gate, nullptr, E_, 0, nullptr, 0);
        // x_t = emb_t + gate * ctx  (bf16)
        xt_kernel<<<dim3(E_ / 256, B_), 256, 0, stream>>>(captions, t, emb, gate,
                                                          ctx, xt_bf);
        // gates = x_t @ Wih^T + bih + h @ Whh^T + bhh  (dual-GEMM fused)
        gemm_wmma<true,false,0,false><<<dim3(1, (4 * D_) / 32), 128, 0, stream>>>(
            xt_bf, Wih_bf, bih, E_, h_bf, Whh_bf, bhh, D_,
            gates, nullptr, 4 * D_, 0, nullptr, 0);
        // LSTM cell -> h (bf16), c (f32)
        lstm_kernel<<<(B_ * D_) / 256, 256, 0, stream>>>(gates, h_bf, c);
        // preds = (h_new @ Wfc^T + bfc) * mask -> straight into d_out
        // 312 full 32-wide tiles (cols 0..9983) + one 16-wide tail (9984..9999)
        gemm_wmma<false,false,0,true><<<dim3(1, (V_ - 16) / 32), 128, 0, stream>>>(
            h_bf, Wfc_bf, bfc, D_, nullptr, nullptr, nullptr, 0,
            out_pred + (size_t)t * V_, nullptr, T * V_, 0, caplen, t);
        gemm_wmma<false,true,0,true><<<dim3(1, 1), 128, 0, stream>>>(
            h_bf, Wfc_bf, bfc, D_, nullptr, nullptr, nullptr, 0,
            out_pred + (size_t)t * V_, nullptr, T * V_, V_ - 16, caplen, t);
    }
}
